// GraphAttentionLayer_86535001079907
// MI455X (gfx1250) — compile-verified, matrix operand order checked by reading, not run on hardware
//
#include <hip/hip_runtime.h>
#include <hip/hip_bf16.h>

#define N     8192
#define FIN   128
#define FOUT  64
#define SPLIT 8

typedef __attribute__((ext_vector_type(16))) _Float16 v16h;
typedef __attribute__((ext_vector_type(8)))  _Float16 v8h;
typedef __attribute__((ext_vector_type(8)))  float    v8f;
typedef __attribute__((ext_vector_type(4)))  float    v4f;

__device__ __forceinline__ unsigned enc_ord(float f) {
    unsigned u = __float_as_uint(f);
    return (u & 0x80000000u) ? ~u : (u | 0x80000000u);
}
__device__ __forceinline__ float dec_ord(unsigned u) {
    return (u & 0x80000000u) ? __uint_as_float(u ^ 0x80000000u)
                             : __uint_as_float(~u);
}

// ---------------- Kernel 1: Wh = x @ W ; also WhT (f16, transposed) ----------
__global__ void k_wh(const float* __restrict__ x, const float* __restrict__ W,
                     float* __restrict__ Wh, _Float16* __restrict__ WhT) {
    __shared__ float Ws[FIN * FOUT];            // 32 KB
    const int tid = threadIdx.x;
    for (int i = tid; i < FIN * FOUT; i += 256) Ws[i] = W[i];
    __syncthreads();
    const int row = blockIdx.x * 4 + (tid >> 6);
    const int c   = tid & 63;
    const float* xr = x + (size_t)row * FIN;
    float acc = 0.f;
#pragma unroll 8
    for (int k = 0; k < FIN; ++k) acc = fmaf(xr[k], Ws[k * FOUT + c], acc);
    Wh [(size_t)row * FOUT + c] = acc;
    WhT[(size_t)c   * N    + row] = (_Float16)acc;
}

// ---------------- Kernel 2: Wh1, Wh2, global max(Wh2) ------------------------
__global__ void k_vec(const float* __restrict__ Wh, const float* __restrict__ a,
                      float* __restrict__ Wh1, float* __restrict__ Wh2,
                      unsigned* __restrict__ maxbits) {
    const int i = blockIdx.x * 256 + threadIdx.x;
    const float* r = Wh + (size_t)i * FOUT;
    float s1 = 0.f, s2 = 0.f;
#pragma unroll 8
    for (int c = 0; c < FOUT; ++c) {
        float v = r[c];
        s1 = fmaf(v, a[c], s1);
        s2 = fmaf(v, a[FOUT + c], s2);
    }
    Wh1[i] = s1;
    Wh2[i] = s2;
    unsigned e = enc_ord(s2);
#pragma unroll
    for (int off = 16; off > 0; off >>= 1) {
        unsigned o = __shfl_xor(e, off, 32);
        e = (o > e) ? o : e;
    }
    if ((threadIdx.x & 31) == 0) atomicMax(maxbits, e);
}

// ---------------- Kernel 3: fused masked-softmax @ Wh (flash style, WMMA) ----
// One wave per (16-row tile, j-split). A = p tile (16x32 f16, built in regs),
// B = WhT tile (32x16 f16 x4), C/D = f32 16x16 x4 accumulators.
// Fully software-pipelined: ALL loads for iteration i+1 are issued before the
// compute of iteration i, so the compiler waits with loadcnt>0 and the HBM
// latency of the adj stream overlaps exp/WMMA work.

struct Tile {
    v4f A0, A1, A2, A3;          // adj (streaming)
    v4f W0, W1, W2, W3;          // Wh2 (L2-hot)
    v8h b0l, b0h, b1l, b1h, b2l, b2h, b3l, b3h;  // WhT B-fragments (L2-hot)
};

__global__ void __launch_bounds__(128)
k_flash(const float* __restrict__ adj, const _Float16* __restrict__ whT,
        const float* __restrict__ Wh1, const float* __restrict__ Wh2,
        const unsigned* __restrict__ maxbits,
        float* __restrict__ accP, float* __restrict__ lP) {
    const int lane = threadIdx.x & 31;
    const int wv   = threadIdx.x >> 5;
    const int rowtile = blockIdx.x * 4 + wv;
    const int split   = blockIdx.y;
    const int i0  = rowtile * 16;
    const int n   = lane & 15;      // output col within 16-frag / A row
    const int hi  = lane >> 4;      // lane half selects K groups
    const int kb1 = hi * 8;         // K = kb1..kb1+7
    const int kb2 = 16 + hi * 8;    // K = kb2..kb2+7
    const int row = i0 + n;

    const float wh1r  = Wh1[row];
    const float maxw2 = dec_ord(*maxbits);
    const float tt = wh1r + maxw2;
    const float m  = tt > 0.f ? tt : 0.2f * tt;   // fixed per-row softmax bound

    v8f acc0 = {}, acc1 = {}, acc2 = {}, acc3 = {};
    float s0 = 0.f, s1 = 0.f, s2 = 0.f, s3 = 0.f;

    const int jbeg = split * (N / SPLIT);
    const int jend = jbeg + (N / SPLIT);
    const float* arow = adj + (size_t)row * N;
    const _Float16* bcol = whT + (size_t)n * N;   // col n of fragment group 0

    auto load_tile = [&](int j0) -> Tile {
        Tile t;
        t.A0 = __builtin_nontemporal_load((const v4f*)(arow + j0 + kb1));
        t.A1 = __builtin_nontemporal_load((const v4f*)(arow + j0 + kb1 + 4));
        t.A2 = __builtin_nontemporal_load((const v4f*)(arow + j0 + kb2));
        t.A3 = __builtin_nontemporal_load((const v4f*)(arow + j0 + kb2 + 4));
        t.W0 = *(const v4f*)(Wh2 + j0 + kb1);
        t.W1 = *(const v4f*)(Wh2 + j0 + kb1 + 4);
        t.W2 = *(const v4f*)(Wh2 + j0 + kb2);
        t.W3 = *(const v4f*)(Wh2 + j0 + kb2 + 4);
        const _Float16* c0 = bcol + j0;
        t.b0l = *(const v8h*)(c0 + kb1);
        t.b0h = *(const v8h*)(c0 + kb2);
        const _Float16* c1 = c0 + (size_t)16 * N;
        t.b1l = *(const v8h*)(c1 + kb1);
        t.b1h = *(const v8h*)(c1 + kb2);
        const _Float16* c2 = c0 + (size_t)32 * N;
        t.b2l = *(const v8h*)(c2 + kb1);
        t.b2h = *(const v8h*)(c2 + kb2);
        const _Float16* c3 = c0 + (size_t)48 * N;
        t.b3l = *(const v8h*)(c3 + kb1);
        t.b3h = *(const v8h*)(c3 + kb2);
        return t;
    };

    auto compute = [&](const Tile& t) {
        v16h af;
#pragma unroll
        for (int k = 0; k < 4; ++k) {
            float e0, e, p;
            e0 = wh1r + t.W0[k]; e = e0 > 0.f ? e0 : 0.2f * e0;
            p  = t.A0[k] * __builtin_amdgcn_exp2f((e - m) * 1.44269504f);
            s0 += p; af[k] = (_Float16)p;
            e0 = wh1r + t.W1[k]; e = e0 > 0.f ? e0 : 0.2f * e0;
            p  = t.A1[k] * __builtin_amdgcn_exp2f((e - m) * 1.44269504f);
            s1 += p; af[4 + k] = (_Float16)p;
            e0 = wh1r + t.W2[k]; e = e0 > 0.f ? e0 : 0.2f * e0;
            p  = t.A2[k] * __builtin_amdgcn_exp2f((e - m) * 1.44269504f);
            s2 += p; af[8 + k] = (_Float16)p;
            e0 = wh1r + t.W3[k]; e = e0 > 0.f ? e0 : 0.2f * e0;
            p  = t.A3[k] * __builtin_amdgcn_exp2f((e - m) * 1.44269504f);
            s3 += p; af[12 + k] = (_Float16)p;
        }
        v16h bf;
        bf = __builtin_shufflevector(t.b0l, t.b0h, 0,1,2,3,4,5,6,7,8,9,10,11,12,13,14,15);
        acc0 = __builtin_amdgcn_wmma_f32_16x16x32_f16(false, af, false, bf,
                                                      (short)0, acc0, false, false);
        bf = __builtin_shufflevector(t.b1l, t.b1h, 0,1,2,3,4,5,6,7,8,9,10,11,12,13,14,15);
        acc1 = __builtin_amdgcn_wmma_f32_16x16x32_f16(false, af, false, bf,
                                                      (short)0, acc1, false, false);
        bf = __builtin_shufflevector(t.b2l, t.b2h, 0,1,2,3,4,5,6,7,8,9,10,11,12,13,14,15);
        acc2 = __builtin_amdgcn_wmma_f32_16x16x32_f16(false, af, false, bf,
                                                      (short)0, acc2, false, false);
        bf = __builtin_shufflevector(t.b3l, t.b3h, 0,1,2,3,4,5,6,7,8,9,10,11,12,13,14,15);
        acc3 = __builtin_amdgcn_wmma_f32_16x16x32_f16(false, af, false, bf,
                                                      (short)0, acc3, false, false);
    };

    // software pipeline: loads for i+1 issued before compute of i
    Tile cur = load_tile(jbeg);
#pragma unroll 1
    for (int j0 = jbeg; j0 < jend - 32; j0 += 32) {
        Tile nxt = load_tile(j0 + 32);
        compute(cur);
        cur = nxt;
    }
    compute(cur);

    // write partial accumulators: C layout VGPR k -> row k + 8*hi, col = lane&15
    float* ap = accP + ((size_t)split * N + i0) * FOUT;
#pragma unroll
    for (int k = 0; k < 8; ++k) {
        const size_t r = (size_t)(k + 8 * hi) * FOUT;
        ap[r +  0 + n] = acc0[k];
        ap[r + 16 + n] = acc1[k];
        ap[r + 32 + n] = acc2[k];
        ap[r + 48 + n] = acc3[k];
    }
    // per-row partial denominator: lane l and l+16 together cover all 32 K
    float s = (s0 + s1) + (s2 + s3);
    s += __shfl_xor(s, 16, 32);
    if (lane < 16) lP[(size_t)split * N + i0 + lane] = s;
}

// ---------------- Kernel 4: combine splits, normalize, ELU -------------------
__global__ void k_final(const float* __restrict__ accP, const float* __restrict__ lP,
                        float* __restrict__ out) {
    const int idx = blockIdx.x * 256 + threadIdx.x;
    const int i = idx >> 6;
    const int c = idx & 63;
    float av = 0.f, lv = 0.f;
#pragma unroll
    for (int sp = 0; sp < SPLIT; ++sp) {
        av += accP[((size_t)sp * N + i) * FOUT + c];
        lv += lP[(size_t)sp * N + i];
    }
    const float r = av / lv;
    out[idx] = r > 0.f ? r : (__builtin_amdgcn_exp2f(r * 1.44269504f) - 1.f);
}

// ---------------- launch -----------------------------------------------------
extern "C" void kernel_launch(void* const* d_in, const int* in_sizes, int n_in,
                              void* d_out, int out_size, void* d_ws, size_t ws_size,
                              hipStream_t stream) {
    const float* x   = (const float*)d_in[0];   // [N, FIN]
    const float* adj = (const float*)d_in[1];   // [N, N]
    const float* W   = (const float*)d_in[2];   // [FIN, FOUT]
    const float* a   = (const float*)d_in[3];   // [2*FOUT, 1]
    float* out = (float*)d_out;                 // [N, FOUT]

    char* ws = (char*)d_ws;
    float*     Wh      = (float*)ws;                                      // 2 MB
    _Float16*  WhT     = (_Float16*)(ws + (size_t)2 * 1024 * 1024);       // 1 MB
    float*     Wh1     = (float*)(ws + (size_t)3 * 1024 * 1024);          // 32 KB
    float*     Wh2     = (float*)(ws + (size_t)3 * 1024 * 1024 + 32768);  // 32 KB
    unsigned*  maxbits = (unsigned*)(ws + (size_t)3 * 1024 * 1024 + 65536);
    float*     accP    = (float*)(ws + (size_t)3 * 1024 * 1024 + 65536 + 256); // SPLIT*2MB
    float*     lP      = accP + (size_t)SPLIT * N * FOUT;                 // SPLIT*32KB

    hipMemsetAsync(maxbits, 0, sizeof(unsigned), stream);

    k_wh  <<<N / 4, 256, 0, stream>>>(x, W, Wh, WhT);
    k_vec <<<N / 256, 256, 0, stream>>>(Wh, a, Wh1, Wh2, maxbits);

    dim3 grid(N / 16 / 4, SPLIT);               // 128 x 8 blocks, 4 waves each
    k_flash<<<grid, 128, 0, stream>>>(adj, WhT, Wh1, Wh2, maxbits, accP, lP);

    k_final<<<(N * FOUT) / 256, 256, 0, stream>>>(accP, lP, out);
}
